// PerAttrAttention_6975026889405
// MI455X (gfx1250) — matrix operand order repaired and checked by reading, not compiled, not run on hardware
//
#include <hip/hip_runtime.h>
#include <hip/hip_bf16.h>

// Problem constants (match reference)
#define BATCH   64
#define SEQ     2048
#define HDIM    512
#define NATTR   16
#define NEG_INF (-1e9f)
#define SCALE   0.044194173824159216f   // 512^-0.5

typedef __attribute__((ext_vector_type(16))) __bf16 v16bf;
typedef __attribute__((ext_vector_type(8)))  float  v8f;

// float -> bf16, round-to-nearest-even
__device__ __forceinline__ __bf16 f2bf(float f) {
    unsigned u = __builtin_bit_cast(unsigned, f);
    unsigned r = (u + 0x7FFFu + ((u >> 16) & 1u)) >> 16;
    return __builtin_bit_cast(__bf16, (unsigned short)r);
}

// A-fragment K base for 16-bit 16x32 A layout (pair p, lane half)
__device__ __forceinline__ int a_kbase(int p, int half) {
    return (p < 4) ? ((half ? 8 : 0) + 2 * p)
                   : ((half ? 24 : 16) + 2 * (p - 4));
}
// B-fragment K base for 16-bit 32x16 B layout (pair p, lane half)
__device__ __forceinline__ int b_kbase(int p, int half) {
    return (p < 4) ? ((half ? 8 : 0) + 2 * p)
                   : ((half ? 24 : 16) + 2 * (p - 4));
}

// ---------------------------------------------------------------------------
// Kernel 1: scores[b, n, t] = scale * <hidden[b,t,:], queries[n,:]>, masked.
// One wave computes one 16(attr) x 16(t) tile. Raw masked scores go straight
// into the weights region of d_out (kernel 2 softmaxes in place).
// Grid: (BATCH, SEQ/64), block: 128 (4 waves).
// ---------------------------------------------------------------------------
__global__ void scores_kernel(const float* __restrict__ hidden,
                              const int*   __restrict__ pad_mask,
                              const float* __restrict__ queries,
                              float*       __restrict__ scores_out) {
    const int lane = threadIdx.x & 31;
    const int wave = threadIdx.x >> 5;
    const int half = lane >> 4;
    const int col  = lane & 15;

    const int b  = blockIdx.x;
    const int t0 = (blockIdx.y * 4 + wave) * 16;
    const int t  = t0 + col;

    const float* hb = hidden + (size_t)b * SEQ * HDIM;

    v8f acc = {};
    for (int h0 = 0; h0 < HDIM; h0 += 32) {
        v16bf afrag, bfrag;
        // A = queries[m=col, h0+k]  (row M striped across VGPRs in one lane)
        const float* qp = queries + (size_t)col * HDIM + h0;
#pragma unroll
        for (int p = 0; p < 8; ++p) {
            int kb = a_kbase(p, half);
            afrag[2 * p]     = f2bf(qp[kb]);
            afrag[2 * p + 1] = f2bf(qp[kb + 1]);
        }
        // B[k, n=col] = hidden[b, t0+col, h0+k]
        const float* hp = hb + (size_t)t * HDIM + h0;
        if (h0 + 32 < HDIM) __builtin_prefetch(hp + 32, 0, 1);
#pragma unroll
        for (int p = 0; p < 8; ++p) {
            int kb = b_kbase(p, half);
            bfrag[2 * p]     = f2bf(hp[kb]);
            bfrag[2 * p + 1] = f2bf(hp[kb + 1]);
        }
        acc = __builtin_amdgcn_wmma_f32_16x16x32_bf16(
                  false, afrag, false, bfrag, (short)0, acc, false, false);
    }

    const bool masked = pad_mask[(size_t)b * SEQ + t] != 0;
#pragma unroll
    for (int v = 0; v < 8; ++v) {
        const int m = v + 8 * half;            // attr index
        float s = masked ? NEG_INF : acc[v] * SCALE;
        scores_out[((size_t)(b * NATTR + m)) * SEQ + t] = s;
    }
}

// ---------------------------------------------------------------------------
// Kernel 2: in-place softmax over T for each of the B*NATTR rows.
// Grid: B*NATTR blocks, 256 threads; each thread owns 8 elements.
// ---------------------------------------------------------------------------
__global__ void softmax_kernel(float* __restrict__ w) {
    __shared__ float red[256];
    const int tid = threadIdx.x;
    float* row = w + (size_t)blockIdx.x * SEQ;

    float local[8];
    float mx = -__builtin_inff();
#pragma unroll
    for (int i = 0; i < 8; ++i) {
        local[i] = row[tid + i * 256];
        mx = fmaxf(mx, local[i]);
    }
    red[tid] = mx;
    __syncthreads();
    for (int s = 128; s > 0; s >>= 1) {
        if (tid < s) red[tid] = fmaxf(red[tid], red[tid + s]);
        __syncthreads();
    }
    mx = red[0];
    __syncthreads();

    float sum = 0.0f;
#pragma unroll
    for (int i = 0; i < 8; ++i) {
        local[i] = __expf(local[i] - mx);
        sum += local[i];
    }
    red[tid] = sum;
    __syncthreads();
    for (int s = 128; s > 0; s >>= 1) {
        if (tid < s) red[tid] += red[tid + s];
        __syncthreads();
    }
    const float inv = 1.0f / red[0];
#pragma unroll
    for (int i = 0; i < 8; ++i)
        row[tid + i * 256] = local[i] * inv;
}

// ---------------------------------------------------------------------------
// Kernel 3: context[b, n, h] = sum_t weights[b,n,t] * hidden[b,t,h].
// One wave computes one 16(attr) x 16(h) tile, K-loop over T in steps of 32.
// Grid: (BATCH, HDIM/64), block: 128 (4 waves).
// ---------------------------------------------------------------------------
__global__ void context_kernel(const float* __restrict__ hidden,
                               const float* __restrict__ weights,
                               float*       __restrict__ ctx) {
    const int lane = threadIdx.x & 31;
    const int wave = threadIdx.x >> 5;
    const int half = lane >> 4;
    const int col  = lane & 15;

    const int b  = blockIdx.x;
    const int h0 = (blockIdx.y * 4 + wave) * 16;

    const float* hb = hidden + (size_t)b * SEQ * HDIM;

    v8f acc = {};
    for (int kt0 = 0; kt0 < SEQ; kt0 += 32) {
        v16bf afrag, bfrag;
        // A = weights[b, m=col, kt0+k]
        const float* wp = weights + ((size_t)(b * NATTR + col)) * SEQ + kt0;
#pragma unroll
        for (int p = 0; p < 8; ++p) {
            int kb = a_kbase(p, half);
            afrag[2 * p]     = f2bf(wp[kb]);
            afrag[2 * p + 1] = f2bf(wp[kb + 1]);
        }
        // B[k, n=col] = hidden[b, kt0+k, h0+col]  (stride HDIM along k)
        const float* hp = hb + (size_t)kt0 * HDIM + h0 + col;
        if (kt0 + 32 < SEQ) __builtin_prefetch(hp + (size_t)32 * HDIM, 0, 1);
#pragma unroll
        for (int p = 0; p < 8; ++p) {
            int kb = b_kbase(p, half);
            bfrag[2 * p]     = f2bf(hp[(size_t)kb * HDIM]);
            bfrag[2 * p + 1] = f2bf(hp[(size_t)(kb + 1) * HDIM]);
        }
        acc = __builtin_amdgcn_wmma_f32_16x16x32_bf16(
                  false, afrag, false, bfrag, (short)0, acc, false, false);
    }

#pragma unroll
    for (int v = 0; v < 8; ++v) {
        const int m = v + 8 * half;            // attr index
        ctx[((size_t)(b * NATTR + m)) * HDIM + h0 + col] = acc[v];
    }
}

// ---------------------------------------------------------------------------
extern "C" void kernel_launch(void* const* d_in, const int* in_sizes, int n_in,
                              void* d_out, int out_size, void* d_ws, size_t ws_size,
                              hipStream_t stream) {
    const float* hidden   = (const float*)d_in[0];
    const int*   pad_mask = (const int*)d_in[1];
    const float* queries  = (const float*)d_in[2];

    float* out = (float*)d_out;
    float* ctx = out;                                   // [B, NATTR, HDIM]
    float* wts = out + (size_t)BATCH * NATTR * HDIM;    // [B, NATTR, SEQ]

    // 1) masked, scaled scores -> weights region (WMMA)
    scores_kernel<<<dim3(BATCH, SEQ / 64), 128, 0, stream>>>(
        hidden, pad_mask, queries, wts);

    // 2) in-place row softmax
    softmax_kernel<<<BATCH * NATTR, 256, 0, stream>>>(wts);

    // 3) context = weights x hidden (WMMA)
    context_kernel<<<dim3(BATCH, HDIM / 64), 128, 0, stream>>>(
        hidden, wts, ctx);
}
